// GAT_4879082848620
// MI455X (gfx1250) — compile-verified, matrix operand order checked by reading, not run on hardware
//
#include <hip/hip_runtime.h>
#include <hip/hip_bf16.h>

// ---------------------------------------------------------------------------
// GAT 2-layer forward for MI455X (gfx1250, wave32).
// GEMMs: LDS-staged weight panel (f32->f16), 16 static v_wmma_f32_16x16x32_f16
// per wave (16x64 output strip, A-fragment reuse). Attention softmax /
// aggregation via L2-resident float atomics.
// ---------------------------------------------------------------------------

typedef __attribute__((ext_vector_type(16))) _Float16 v16h;
typedef __attribute__((ext_vector_type(8)))  _Float16 v8h;
typedef __attribute__((ext_vector_type(8)))  float    v8f;

#define HEADS 8
#define NEG_SLOPE 0.2f
#define KDIM 128          // inner dim of both GEMMs
#define NSTRIP 64         // output columns per block panel / per wave strip

// Element j of a 16-bit A/B fragment maps to K offset (ISA 7.12.2):
// j in [0,8): k = lhalf*8 + j ; j in [8,16): k = 16 + lhalf*8 + (j-8).

// ------------------------------ GEMM kernel --------------------------------
// C[M,Nout] = A[M,KDIM] x B[KDIM,Nout].
// Block: 256 threads (8 waves). blockIdx.y selects 64-col panel; B panel is
// staged to LDS as f16 in [n][k] layout (fragment runs are 16B-contiguous).
// Each wave computes a 16x64 strip: 4 accumulators, 4 K-steps, 16 WMMAs.
__global__ void __launch_bounds__(256)
gat_gemm_f16(const float* __restrict__ A,
             const float* __restrict__ B,
             float* __restrict__ C,
             int M, int Nout) {
    __shared__ _Float16 ldsB[NSTRIP * KDIM];   // 16 KB

    const int tid  = threadIdx.x;
    const int wave = tid >> 5;
    const int lane = tid & 31;
    const int n0   = blockIdx.y * NSTRIP;

    // cooperative stage: B[k][n0+n] (f32, coalesced) -> ldsB[n*KDIM+k] (f16)
    for (int idx = tid; idx < NSTRIP * KDIM; idx += 256) {
        int k = idx >> 6;      // 0..127
        int n = idx & 63;      // 0..63
        ldsB[n * KDIM + k] = (_Float16)B[(size_t)k * Nout + (n0 + n)];
    }
    __syncthreads();

    const int mtile = blockIdx.x * 8 + wave;
    if (mtile >= (M >> 4)) return;             // wave-uniform guard
    const int m0 = mtile << 4;

    const int lhalf = lane >> 4;
    const int r     = lane & 15;
    const float* arow = A + (size_t)(m0 + r) * KDIM;

    v8f acc[4] = {};
#pragma unroll
    for (int k0 = 0; k0 < KDIM; k0 += 32) {
        if (k0 + 32 < KDIM)                     // global_prefetch_b8
            __builtin_prefetch(arow + k0 + 32, 0, 3);
        // A fragment: two contiguous 8-float runs, f32->f16 in regs
        v16h a;
#pragma unroll
        for (int j = 0; j < 8; ++j) {
            a[j]     = (_Float16)arow[k0 + lhalf * 8 + j];
            a[j + 8] = (_Float16)arow[k0 + 16 + lhalf * 8 + j];
        }
        // 4 B fragments from LDS; each is two 16-byte ds_load_b128 runs
#pragma unroll
        for (int nt = 0; nt < 4; ++nt) {
            const _Float16* col = ldsB + (size_t)(nt * 16 + r) * KDIM;
            v8h lo = *(const v8h*)(col + k0 + lhalf * 8);
            v8h hi = *(const v8h*)(col + k0 + 16 + lhalf * 8);
            v16h b = __builtin_shufflevector(lo, hi,
                         0, 1, 2, 3, 4, 5, 6, 7, 8, 9, 10, 11, 12, 13, 14, 15);
            acc[nt] = __builtin_amdgcn_wmma_f32_16x16x32_f16(
                          false, a, false, b, (short)0, acc[nt], false, false);
        }
    }

    // C layout (ISA): lane half selects M bias (+8), VGPR v -> row m0+v
#pragma unroll
    for (int nt = 0; nt < 4; ++nt)
#pragma unroll
        for (int v = 0; v < 8; ++v)
            C[(size_t)(m0 + lhalf * 8 + v) * Nout + (n0 + nt * 16 + r)] = acc[nt][v];
}

// --------------------------- small utility kernels -------------------------
__global__ void fill_f32(float* __restrict__ p, float v, long long n) {
    long long i = (long long)blockIdx.x * blockDim.x + threadIdx.x;
    if (i < n) p[i] = v;
}
__global__ void fill_u32(unsigned* __restrict__ p, unsigned v, long long n) {
    long long i = (long long)blockIdx.x * blockDim.x + threadIdx.x;
    if (i < n) p[i] = v;
}

// order-preserving float<->uint encoding for atomicMax-based segment max
__device__ __forceinline__ unsigned enc_f32(float x) {
    unsigned u = __float_as_uint(x);
    return (u & 0x80000000u) ? ~u : (u | 0x80000000u);
}
__device__ __forceinline__ float dec_f32(unsigned u) {
    u = (u & 0x80000000u) ? (u & 0x7FFFFFFFu) : ~u;
    return __uint_as_float(u);
}
__device__ __forceinline__ float leaky(float x) {
    return x > 0.f ? x : NEG_SLOPE * x;
}

// per-(node,head) attention logits: as = <h_nh, attS_h>, ad = <h_nh, attD_h>
__global__ void alpha_kernel(const float* __restrict__ feat,
                             const float* __restrict__ attS,
                             const float* __restrict__ attD,
                             float* __restrict__ as, float* __restrict__ ad,
                             int N, int D) {
    int t = blockIdx.x * blockDim.x + threadIdx.x;
    if (t >= N * HEADS) return;
    int n = t >> 3, h = t & 7;
    const float* f = feat + (size_t)n * (HEADS * D) + h * D;
    float s = 0.f, d = 0.f;
    for (int i = 0; i < D; ++i) {
        s += f[i] * attS[h * D + i];
        d += f[i] * attD[h * D + i];
    }
    as[t] = s; ad[t] = d;
}

__device__ __forceinline__ void edge_src_dst(const long long* __restrict__ ei,
                                             int e, int E, long long& s, long long& d) {
    if (e < E) { s = ei[e]; d = ei[E + e]; }
    else       { s = e - E; d = e - E; }     // synthesized self-loop
}

// pass 1: segment max over incoming edges (encoded uint atomicMax)
__global__ void edge_max_kernel(const long long* __restrict__ ei,
                                const float* __restrict__ as,
                                const float* __restrict__ ad,
                                unsigned* __restrict__ menc, int E, int ET) {
    int e = blockIdx.x * blockDim.x + threadIdx.x;
    if (e >= ET) return;
    long long s, d;
    edge_src_dst(ei, e, E, s, d);
#pragma unroll
    for (int h = 0; h < HEADS; ++h) {
        float v = leaky(as[s * HEADS + h] + ad[d * HEADS + h]);
        atomicMax(&menc[d * HEADS + h], enc_f32(v));
    }
}

// decode max in place (uint -> float), non-finite -> 0 (matches reference)
__global__ void decode_max_kernel(unsigned* __restrict__ menc, long long n) {
    long long i = (long long)blockIdx.x * blockDim.x + threadIdx.x;
    if (i >= n) return;
    float m = dec_f32(menc[i]);
    if (!(fabsf(m) < 3.0e38f)) m = 0.f;     // catches NaN and +-inf
    ((float*)menc)[i] = m;
}

// pass 2: denom = segment_sum(exp(e - m[dst]))
__global__ void edge_sum_kernel(const long long* __restrict__ ei,
                                const float* __restrict__ as,
                                const float* __restrict__ ad,
                                const float* __restrict__ mflt,
                                float* __restrict__ denom, int E, int ET) {
    int e = blockIdx.x * blockDim.x + threadIdx.x;
    if (e >= ET) return;
    long long s, d;
    edge_src_dst(ei, e, E, s, d);
#pragma unroll
    for (int h = 0; h < HEADS; ++h) {
        float v  = leaky(as[s * HEADS + h] + ad[d * HEADS + h]);
        float ex = __expf(v - mflt[d * HEADS + h]);
        atomicAdd(&denom[d * HEADS + h], ex);
    }
}

// pass 3: agg[dst] += alpha * feat[src]; one thread per (edge, head)
__global__ void edge_agg_kernel(const long long* __restrict__ ei,
                                const float* __restrict__ as,
                                const float* __restrict__ ad,
                                const float* __restrict__ mflt,
                                const float* __restrict__ denom,
                                const float* __restrict__ feat,
                                float* __restrict__ agg,
                                int E, int ET, int D) {
    long long t = (long long)blockIdx.x * blockDim.x + threadIdx.x;
    if (t >= (long long)ET * HEADS) return;
    int e = (int)(t >> 3), h = (int)(t & 7);
    long long s, d;
    edge_src_dst(ei, e, E, s, d);
    float v     = leaky(as[s * HEADS + h] + ad[d * HEADS + h]);
    float ex    = __expf(v - mflt[d * HEADS + h]);
    float alpha = ex / (denom[d * HEADS + h] + 1e-16f);
    const float* fs = feat + (size_t)s * (HEADS * D) + h * D;
    float*       ag = agg  + (size_t)d * (HEADS * D) + h * D;
    for (int i = 0; i < D; ++i)
        atomicAdd(&ag[i], fs[i] * alpha);
}

// layer-1 finalize: in-place bias + ReLU on concat layout [N, 128]
__global__ void finalize1_kernel(float* __restrict__ agg,
                                 const float* __restrict__ bias, long long n) {
    long long i = (long long)blockIdx.x * blockDim.x + threadIdx.x;
    if (i >= n) return;
    float v = agg[i] + bias[i & 127];
    agg[i] = v > 0.f ? v : 0.f;
}

// layer-2 finalize: mean over 8 heads + bias -> d_out [N, 64]
__global__ void finalize2_kernel(const float* __restrict__ agg,
                                 const float* __restrict__ bias,
                                 float* __restrict__ out, int N) {
    int t = blockIdx.x * blockDim.x + threadIdx.x;
    if (t >= N * 64) return;
    int n = t >> 6, j = t & 63;
    const float* a = agg + (size_t)n * 512 + j;
    float s = 0.f;
#pragma unroll
    for (int h = 0; h < HEADS; ++h) s += a[h * 64];
    out[t] = s * 0.125f + bias[j];
}

// ------------------------------- launcher ----------------------------------
static inline int cdiv(long long a, int b) { return (int)((a + b - 1) / b); }

extern "C" void kernel_launch(void* const* d_in, const int* in_sizes, int n_in,
                              void* d_out, int out_size, void* d_ws, size_t ws_size,
                              hipStream_t stream) {
    const float*     x   = (const float*)d_in[0];
    const long long* ei  = (const long long*)d_in[1];
    const float*     W1  = (const float*)d_in[2];
    const float*     aS1 = (const float*)d_in[3];
    const float*     aD1 = (const float*)d_in[4];
    const float*     b1  = (const float*)d_in[5];
    const float*     W2  = (const float*)d_in[6];
    const float*     aS2 = (const float*)d_in[7];
    const float*     aD2 = (const float*)d_in[8];
    const float*     b2  = (const float*)d_in[9];
    float*           out = (float*)d_out;

    const int N  = in_sizes[0] / 128;   // 50000
    const int E  = in_sizes[1] / 2;     // 800000
    const int ET = E + N;               // + self loops

    // workspace layout (floats)
    float* ws = (float*)d_ws;
    size_t off = 0;
    float* h1    = ws + off; off += (size_t)N * 128;   // X @ W1
    float* agg1  = ws + off; off += (size_t)N * 128;   // layer-1 agg -> h1_act
    float* h2    = ws + off; off += (size_t)N * 512;   // h1_act @ W2
    float* agg2  = ws + off; off += (size_t)N * 512;   // layer-2 agg
    float* as1   = ws + off; off += (size_t)N * HEADS;
    float* ad1   = ws + off; off += (size_t)N * HEADS;
    float* as2   = ws + off; off += (size_t)N * HEADS;
    float* ad2   = ws + off; off += (size_t)N * HEADS;
    unsigned* menc = (unsigned*)(ws + off); off += (size_t)N * HEADS; // uint->float in place
    float* denom = ws + off; off += (size_t)N * HEADS;

    const int T = 256;
    const long long NH  = (long long)N * HEADS;
    const long long ETH = (long long)ET * HEADS;
    const int mblocks = cdiv(N >> 4, 8);   // 8 M-tiles (waves) per block

    // ---------------- layer 1 ----------------
    {   // GEMM1: h1 = x @ W1   (M=N, K=128, Nout=128)
        dim3 grid(mblocks, 128 / NSTRIP);
        gat_gemm_f16<<<grid, 256, 0, stream>>>(x, W1, h1, N, 128);
    }
    alpha_kernel<<<cdiv(NH, T), T, 0, stream>>>(h1, aS1, aD1, as1, ad1, N, 16);

    fill_u32<<<cdiv(NH, T), T, 0, stream>>>(menc, 0u, NH);
    fill_f32<<<cdiv(NH, T), T, 0, stream>>>(denom, 0.f, NH);
    fill_f32<<<cdiv((long long)N * 128, T), T, 0, stream>>>(agg1, 0.f, (long long)N * 128);

    edge_max_kernel<<<cdiv(ET, T), T, 0, stream>>>(ei, as1, ad1, menc, E, ET);
    decode_max_kernel<<<cdiv(NH, T), T, 0, stream>>>(menc, NH);
    edge_sum_kernel<<<cdiv(ET, T), T, 0, stream>>>(ei, as1, ad1, (const float*)menc,
                                                   denom, E, ET);
    edge_agg_kernel<<<cdiv(ETH, T), T, 0, stream>>>(ei, as1, ad1, (const float*)menc,
                                                    denom, h1, agg1, E, ET, 16);
    finalize1_kernel<<<cdiv((long long)N * 128, T), T, 0, stream>>>(agg1, b1,
                                                                    (long long)N * 128);

    // ---------------- layer 2 ----------------
    {   // GEMM2: h2 = h1_act @ W2   (M=N, K=128, Nout=512)
        dim3 grid(mblocks, 512 / NSTRIP);
        gat_gemm_f16<<<grid, 256, 0, stream>>>(agg1, W2, h2, N, 512);
    }
    alpha_kernel<<<cdiv(NH, T), T, 0, stream>>>(h2, aS2, aD2, as2, ad2, N, 64);

    fill_u32<<<cdiv(NH, T), T, 0, stream>>>(menc, 0u, NH);
    fill_f32<<<cdiv(NH, T), T, 0, stream>>>(denom, 0.f, NH);
    fill_f32<<<cdiv((long long)N * 512, T), T, 0, stream>>>(agg2, 0.f, (long long)N * 512);

    edge_max_kernel<<<cdiv(ET, T), T, 0, stream>>>(ei, as2, ad2, menc, E, ET);
    decode_max_kernel<<<cdiv(NH, T), T, 0, stream>>>(menc, NH);
    edge_sum_kernel<<<cdiv(ET, T), T, 0, stream>>>(ei, as2, ad2, (const float*)menc,
                                                   denom, E, ET);
    edge_agg_kernel<<<cdiv(ETH, T), T, 0, stream>>>(ei, as2, ad2, (const float*)menc,
                                                    denom, h2, agg2, E, ET, 64);
    finalize2_kernel<<<cdiv((long long)N * 64, T), T, 0, stream>>>(agg2, b2, out, N);
}
